// RNN_Net_50096498541013
// MI455X (gfx1250) — matrix operand order repaired and checked by reading, not compiled, not run on hardware
//
#include <hip/hip_runtime.h>

// RNN_Net: h_{t+1} = (1-dt)h_t + dt(tanh(h_t) @ W_hh^T + x_t @ W_ih^T); out = hs @ W_dec^T
// B=64, T=512, DIN=128, DH=512, DOUT=128, dt=0.2
#define DT_STEP 0.2f
#define B_SZ   64
#define T_SZ   512
#define DIN    128
#define DH     512
#define DOUT   128

typedef __attribute__((ext_vector_type(16))) __bf16        v16bf;
typedef __attribute__((ext_vector_type(8)))  float         v8f;
typedef __attribute__((ext_vector_type(4)))  unsigned int  uv4;
typedef __attribute__((ext_vector_type(8)))  int           iv8;
typedef __attribute__((ext_vector_type(4)))  int           iv4;

union BFrag { unsigned int u[8]; uint4 q[2]; v16bf v; };

// fp32 -> bf16 (round to nearest even), low 16 bits
__device__ __forceinline__ unsigned int f2bf(float f) {
    unsigned int u = __float_as_uint(f);
    return (u + 0x7fffu + ((u >> 16) & 1u)) >> 16;
}
__device__ __forceinline__ unsigned int packbf(float lo, float hi) {
    return f2bf(lo) | (f2bf(hi) << 16);
}

// A-fragment kp-in-chunk for VGPR v, lane-half hi (16-bit A 16x32, ISA 7.12.2):
// lanes0-15: kp {0-3, 8-11}; lanes16-31: kp {4-7, 12-15}
__device__ __forceinline__ int kpA(int v, int hi) { return (v & 3) + (hi << 2) + ((v >> 2) << 3); }
// B-fragment kp-in-chunk (same as A): lanes0-15: kp {0-3, 8-11}; lanes16-31: kp {4-7, 12-15}
__device__ __forceinline__ int kpB(int v, int hi) { return (v & 3) + (hi << 2) + ((v >> 2) << 3); }

// ---------------------------------------------------------------------------
// TDM helper: 2D dword copy global -> LDS. n_dw dwords per row (<=65535),
// 'rows' rows, contiguous on both sides. lds_addr is a byte offset from the
// dynamic-LDS base (== LDS 0 here: kernel has no static __shared__).
// D# layout per CDNA5 ISA (group0/group1).
// 6-arg builtin form: (g0 uint32x4, g1 int32x8, g2 int32x4, g3 int32x4,
//                      extra int32x8, cpol i32).
// ---------------------------------------------------------------------------
__device__ __forceinline__ void tdm_load_2d(unsigned int lds_addr, const void* gsrc,
                                            unsigned int n_dw, unsigned int rows) {
    unsigned long long ga = (unsigned long long)(uintptr_t)gsrc;
    uv4 g0;
    g0[0] = 1u;                                                // count=1, user mode
    g0[1] = lds_addr;                                          // lds_addr (bytes)
    g0[2] = (unsigned int)ga;                                  // global_addr[31:0]
    g0[3] = ((unsigned int)(ga >> 32) & 0x1FFFFFFu) | (2u << 30); // addr[56:32] | type=2
    iv8 g1;
    g1[0] = (int)(2u << 16);                                   // data_size=4B
    g1[1] = (int)((n_dw & 0xffffu) << 16);                     // tensor_dim0[15:0]
    g1[2] = (int)((n_dw >> 16) | ((rows & 0xffffu) << 16));    // td0[31:16] | td1[15:0]
    g1[3] = (int)(n_dw << 16);                                 // td1[31:16]=0 | tile_dim0
    g1[4] = (int)rows;                                         // tile_dim1 | tile_dim2=0
    g1[5] = (int)n_dw;                                         // tensor_dim0_stride[31:0]
    g1[6] = (int)((n_dw & 0xffffu) << 16);                     // st0[47:32]=0 | st1[15:0]
    g1[7] = 0;                                                 // st1[47:16]=0
    iv4 z4 = {0, 0, 0, 0};
    iv8 z8 = {0, 0, 0, 0, 0, 0, 0, 0};
    __builtin_amdgcn_tensor_load_to_lds(g0, g1, z4, z4, z8, 0);
}

// ---------------------------------------------------------------------------
// Kernel 0: zero the per-pair split-barrier counters (ws is poisoned once).
// ---------------------------------------------------------------------------
__global__ void zero_ctrs(unsigned int* ctr) {
    if (threadIdx.x < 16) ctr[threadIdx.x] = 0u;
}

// ---------------------------------------------------------------------------
// Kernel 1: repack weights.
//  - W_hh -> full WMMA-B fragment *image*: [half2][nt16][ch16][j2][lane32][dw4]
//    so each fragment is 2 contiguous 512B lane-major slabs (2x ds_load_b128).
//  - W_dec / W_ih -> bf16x2 dwords in B order: pk[kp*N + n] = {W[n][2kp],W[n][2kp+1]}
// ---------------------------------------------------------------------------
#define N_WHH_IMG (2*16*16*2*32*4)   // 131072 dwords
#define N_WDEC_PK (256*128)
#define N_WIH_PK  (64*512)

__global__ __launch_bounds__(256) void pack_weights(
        const float* __restrict__ Wih, const float* __restrict__ Whh,
        const float* __restrict__ Wdec,
        unsigned int* __restrict__ wih_pk, unsigned int* __restrict__ whh_img,
        unsigned int* __restrict__ wdec_pk) {
    int idx = blockIdx.x * 256 + threadIdx.x;
    if (idx < N_WHH_IMG) {
        int dw = idx & 3, ln = (idx >> 2) & 31, j = (idx >> 7) & 1;
        int ch = (idx >> 8) & 15, nt = (idx >> 12) & 15, half = (idx >> 16) & 1;
        int v = j * 4 + dw, hi = ln >> 4, mlo = ln & 15;
        int kp = ch * 16 + kpB(v, hi);             // global k-pair 0..255
        int n  = half * 256 + nt * 16 + mlo;       // global hidden col
        whh_img[idx] = packbf(Whh[n * DH + 2 * kp], Whh[n * DH + 2 * kp + 1]);
    } else if (idx < N_WHH_IMG + N_WDEC_PK) {
        int j = idx - N_WHH_IMG;
        int kp = j >> 7, n = j & 127;
        wdec_pk[j] = packbf(Wdec[n * DH + 2 * kp], Wdec[n * DH + 2 * kp + 1]);
    } else if (idx < N_WHH_IMG + N_WDEC_PK + N_WIH_PK) {
        int j = idx - (N_WHH_IMG + N_WDEC_PK);
        int kp = j >> 9, n = j & 511;
        wih_pk[j] = packbf(Wih[n * DIN + 2 * kp], Wih[n * DIN + 2 * kp + 1]);
    }
}

// ---------------------------------------------------------------------------
// Kernel 2: xin[row][n] = sum_k x[row][k] * W_ih[n][k], rows = B*T = 32768.
// ---------------------------------------------------------------------------
__global__ __launch_bounds__(256) void gemm_xin(
        const float* __restrict__ x, const unsigned int* __restrict__ wih_pk,
        float* __restrict__ xin) {
    const int lane = threadIdx.x & 31;
    const int wave = threadIdx.x >> 5;
    const int tile = blockIdx.x * 8 + wave;        // 2048*32 tiles
    const int mt = tile >> 5, nt = tile & 31;
    const int m0 = mt * 16, n0 = nt * 16;
    const int mlo = lane & 15, hi = lane >> 4;

    v8f acc = {};
    for (int ch = 0; ch < 4; ++ch) {
        BFrag a, b;
#pragma unroll
        for (int v = 0; v < 8; ++v) {
            const int kp = ch * 16 + kpA(v, hi);
            const float* px = x + (size_t)(m0 + mlo) * DIN + 2 * kp;
            a.u[v] = packbf(px[0], px[1]);
            b.u[v] = wih_pk[(size_t)(ch * 16 + kpB(v, hi)) * DH + n0 + mlo];
        }
        acc = __builtin_amdgcn_wmma_f32_16x16x32_bf16(
                false, a.v, false, b.v, (short)0, acc, false, false);
    }
#pragma unroll
    for (int r = 0; r < 8; ++r)
        xin[(size_t)(m0 + r + hi * 8) * DH + n0 + mlo] = acc[r];
}

// ---------------------------------------------------------------------------
// Kernel 3: recurrence. 8 WGs = 4 pairs; pair p owns batch rows p*16..p*16+15,
// WG half q owns hidden units [q*256, q*256+256) (both as N-output and K-source).
// Its 256KB bf16 half of W_hh lives in LDS (TDM-filled, fragment-image layout).
// Per step: publish own-half tanh A-frags (LDS + L2 exch), global-atomic split
// barrier with partner, TDM-pull partner frags into LDS, 256 WMMAs, Euler update.
// ---------------------------------------------------------------------------
#define LDSB_SW   0                      // W_hh image half: 262144 B
#define LDSB_ST   262144                 // A-frag image: 16 chunks * 1KB
#define LDSB_SH   (262144 + 16384)      // h fp32 [256][17]: 17408 B
#define LDS_TOTAL (262144 + 16384 + 17408)
#define HSTR 17

__global__ __launch_bounds__(512) void rnn_recur(
        const float* __restrict__ xin, const unsigned int* __restrict__ whh_img,
        unsigned int* __restrict__ hs_pk, unsigned int* __restrict__ exch,
        unsigned int* __restrict__ ctr) {
    extern __shared__ char smem[];
    unsigned int* sW = (unsigned int*)(smem + LDSB_SW); // [nt16][ch16][j2][lane32][dw4]
    unsigned int* sT = (unsigned int*)(smem + LDSB_ST); // [ch16][j2][lane32][dw4]
    float*        sH = (float*)(smem + LDSB_SH);        // [n_local 256][HSTR]

    const int tid  = threadIdx.x;
    const int bid  = blockIdx.x;
    const int pair = bid >> 1;            // batch tile
    const int half = bid & 1;             // owned hidden half
    const int lane = tid & 31, wave = tid >> 5;
    const int mlo  = lane & 15, hi = lane >> 4;

    for (int i = tid; i < 256 * HSTR; i += 512) sH[i] = 0.0f;

    // Fill LDS W_hh half image: 65536 dwords, one TDM descriptor.
    if (wave == 0) {
        tdm_load_2d(LDSB_SW, whh_img + (size_t)half * 65536, 8192u, 8u);
        __builtin_amdgcn_s_wait_tensorcnt(0);
    }
    __syncthreads();

    for (int t = 0; t < T_SZ; ++t) {
        const int slot = t & 1;
        // ---- phase A1: hs[t] = h_t (pre-update), own kp half, coalesced ----
        for (int d = tid; d < 2048; d += 512) {
            const int m = d >> 7, kpl = d & 127;
            const float h0 = sH[(2 * kpl) * HSTR + m];
            const float h1 = sH[(2 * kpl + 1) * HSTR + m];
            hs_pk[(size_t)((pair * 16 + m) * T_SZ + t) * 256 + half * 128 + kpl] =
                packbf(h0, h1);
        }
        // ---- phase A2: build tanh A-frag image for own K-half ----
        unsigned int* myex = exch + (size_t)(((pair * 2 + slot) * 2) + half) * 2048;
        for (int d = tid; d < 2048; d += 512) {
            const int chl = d >> 8, rem = d & 255;
            const int j = rem >> 7, ln = (rem >> 2) & 31, dw = rem & 3;
            const int v = j * 4 + dw, lhi = ln >> 4, lm = ln & 15;
            const int kpl = chl * 16 + kpA(v, lhi);        // local k-pair 0..127
            const unsigned int pk = packbf(tanhf(sH[(2 * kpl) * HSTR + lm]),
                                           tanhf(sH[(2 * kpl + 1) * HSTR + lm]));
            sT[(half * 8 + chl) * 256 + rem] = pk;         // own chunks in LDS
            myex[d] = pk;                                  // publish for partner
        }
        __threadfence();
        __syncthreads();
        // ---- split barrier with partner WG (global atomic, monotonic) ----
        if (tid == 0) {
            atomicAdd(&ctr[pair], 1u);
            while (__hip_atomic_load(&ctr[pair], __ATOMIC_RELAXED,
                                     __HIP_MEMORY_SCOPE_AGENT) < 2u * (unsigned)(t + 1)) {}
            __threadfence();
        }
        __syncthreads();
        // ---- TDM: pull partner's 8KB A-frag half straight into LDS ----
        if (wave == 0) {
            const unsigned int* pex =
                exch + (size_t)(((pair * 2 + slot) * 2) + (1 - half)) * 2048;
            tdm_load_2d(LDSB_ST + (unsigned)(1 - half) * 8192u, pex, 2048u, 1u);
            __builtin_amdgcn_s_wait_tensorcnt(0);
        }
        __syncthreads();
        // ---- phase B: f = tanh(h) @ W_hh^T + xin ; h = (1-dt)h + dt f ----
        {
            const int nt = wave;                            // local ntile 0..15
            const unsigned int* wb = sW + nt * 4096;        // 16KB per ntile
            v8f acc = {};
            for (int ch = 0; ch < 16; ++ch) {
                BFrag a, b;
                const unsigned int* ap = sT + ch * 256 + lane * 4;
                a.q[0] = *(const uint4*)(ap);
                a.q[1] = *(const uint4*)(ap + 128);
                const unsigned int* bp = wb + ch * 256 + lane * 4;
                b.q[0] = *(const uint4*)(bp);
                b.q[1] = *(const uint4*)(bp + 128);
                acc = __builtin_amdgcn_wmma_f32_16x16x32_bf16(
                        false, a.v, false, b.v, (short)0, acc, false, false);
            }
#pragma unroll
            for (int r = 0; r < 8; ++r) {
                const int m = r + hi * 8;
                const int nl = nt * 16 + mlo;               // local hidden col
                const float xv = xin[(size_t)((pair * 16 + m) * T_SZ + t) * DH +
                                     half * 256 + nl];
                const float hold = sH[nl * HSTR + m];
                sH[nl * HSTR + m] =
                    (1.0f - DT_STEP) * hold + DT_STEP * (acc[r] + xv);
            }
        }
        __syncthreads();
    }
}

// ---------------------------------------------------------------------------
// Kernel 4: decoder, out[row][n] = sum_k hs[row][k] * W_dec[n][k].
// ---------------------------------------------------------------------------
__global__ __launch_bounds__(256) void gemm_dec(
        const unsigned int* __restrict__ hs_pk,
        const unsigned int* __restrict__ wdec_pk, float* __restrict__ out) {
    const int lane = threadIdx.x & 31;
    const int wave = threadIdx.x >> 5;
    const int tile = blockIdx.x * 8 + wave;        // 2048*8 tiles
    const int mt = tile >> 3, nt = tile & 7;
    const int m0 = mt * 16, n0 = nt * 16;
    const int mlo = lane & 15, hi = lane >> 4;

    v8f acc = {};
    for (int ch = 0; ch < 16; ++ch) {
        BFrag a, b;
#pragma unroll
        for (int v = 0; v < 8; ++v) {
            a.u[v] = hs_pk[(size_t)(m0 + mlo) * 256 + ch * 16 + kpA(v, hi)];
            b.u[v] = wdec_pk[(size_t)(ch * 16 + kpB(v, hi)) * DOUT + n0 + mlo];
        }
        acc = __builtin_amdgcn_wmma_f32_16x16x32_bf16(
                false, a.v, false, b.v, (short)0, acc, false, false);
    }
#pragma unroll
    for (int r = 0; r < 8; ++r)
        out[(size_t)(m0 + r + hi * 8) * DOUT + n0 + mlo] = acc[r];
}

// ---------------------------------------------------------------------------
extern "C" void kernel_launch(void* const* d_in, const int* in_sizes, int n_in,
                              void* d_out, int out_size, void* d_ws, size_t ws_size,
                              hipStream_t stream) {
    (void)in_sizes; (void)n_in; (void)out_size; (void)ws_size;
    const float* x    = (const float*)d_in[0];   // [64,512,128]
    const float* Wih  = (const float*)d_in[1];   // [512,128]
    const float* Whh  = (const float*)d_in[2];   // [512,512]
    const float* Wdec = (const float*)d_in[3];   // [128,512]

    char* ws = (char*)d_ws;
    float*        xin      = (float*)(ws);                                  // 64 MB
    unsigned int* hs_pk    = (unsigned int*)(ws + ((size_t)64u << 20));     // 32 MB
    unsigned int* whh_img  = (unsigned int*)(ws + ((size_t)96u << 20));     // 512 KB
    unsigned int* wdec_pk  = (unsigned int*)(ws + ((size_t)96u << 20) + 524288);
    unsigned int* wih_pk   = (unsigned int*)(ws + ((size_t)96u << 20) + 524288 + 131072);
    unsigned int* exch     = (unsigned int*)(ws + ((size_t)96u << 20) + 524288 + 262144);
    unsigned int* ctr      = (unsigned int*)(ws + ((size_t)96u << 20) + 524288 + 262144 + 131072);

    (void)hipFuncSetAttribute((const void*)rnn_recur,
                              hipFuncAttributeMaxDynamicSharedMemorySize, LDS_TOTAL);

    zero_ctrs<<<1, 64, 0, stream>>>(ctr);
    pack_weights<<<768, 256, 0, stream>>>(Wih, Whh, Wdec, wih_pk, whh_img, wdec_pk);
    gemm_xin<<<8192, 256, 0, stream>>>(x, wih_pk, xin);
    rnn_recur<<<8, 512, LDS_TOTAL, stream>>>(xin, whh_img, hs_pk, exch, ctr);
    gemm_dec<<<2048, 256, 0, stream>>>(hs_pk, wdec_pk, (float*)d_out);
}
